// GraphEmbedding_12953621365069
// MI455X (gfx1250) — matrix operand order repaired and checked by reading, not compiled, not run on hardware
//
#include <hip/hip_runtime.h>
#include <hip/hip_bf16.h>

// ---------------------------------------------------------------------------
// MI455X (gfx1250, wave32) GraphNet pipeline.
// All MLPs run on v_wmma_f32_16x16x32_f16 (f32 accumulate); activations are
// gathered/concatenated on the fly into LDS as f16; weights pre-swizzled once
// per call into the WMMA B-fragment lane layout; segment means via f32 atomics.
// ---------------------------------------------------------------------------

typedef __attribute__((ext_vector_type(16))) _Float16 v16h;
typedef __attribute__((ext_vector_type(8)))  _Float16 v8h;
typedef __attribute__((ext_vector_type(8)))  float    v8f;

#define HDIM    32
#define NNODES  50000
#define NEDGES  400000
#define NBATCH  128

// ---------------------------------------------------------------------------
// Fragment helpers (layouts per CDNA5 ISA 7.12.2, wave32)
// ---------------------------------------------------------------------------

__device__ __forceinline__ v16h make_v16(v8h lo, v8h hi) {
  union { v16h v; v8h h[2]; } u;
  u.h[0] = lo; u.h[1] = hi;
  return u.v;
}

// A fragment: 16x32 f16 from a row-major f16 tile in LDS.
// lanes 0-15 hold row M=lane, K = kbase+[0..7] and kbase+[16..23];
// lanes 16-31 hold row M=lane-16, K = kbase+[8..15] and kbase+[24..31].
__device__ __forceinline__ v16h load_a_frag(const _Float16* tile, int ldk,
                                            int kbase, int lane) {
  const int m  = lane & 15;
  const int kh = (lane >> 4) << 3;        // 0 or 8
  const _Float16* p = tile + m * ldk + kbase + kh;
  v8h lo = *(const v8h*)(p);              // ds_load_b128
  v8h hi = *(const v8h*)(p + 16);         // ds_load_b128
  return make_v16(lo, hi);
}

// B fragment: 32x16 f16, pre-swizzled so each lane reads 16 contiguous halfs.
__device__ __forceinline__ v16h load_b_frag(const _Float16* w, int frag, int lane) {
  return *(const v16h*)(w + (size_t)frag * 512 + lane * 16);  // 2x global_load_b128
}

// ---------------------------------------------------------------------------
// Row builders: fuse every gather/concat/product into the LDS staging pass.
// ---------------------------------------------------------------------------

struct EncB {            // [raw(16) | pad]
  const float* p;
  __device__ float operator()(int r, int k) const {
    return k < 16 ? p[r * 16 + k] : 0.f;
  }
};

struct RecEdgeB {        // [x_c[src]-x_c[dst] (48) | e_c (48) | u_c[b[src]] (48) | pad]
  const float *x, *xh, *e, *eh, *u, *uh;
  const int *src, *dst, *batch;
  __device__ float operator()(int r, int k) const {
    if (k < 48) {
      int s = src[r], d = dst[r];
      return k < 16 ? x[s * 16 + k] - x[d * 16 + k]
                    : xh[s * HDIM + (k - 16)] - xh[d * HDIM + (k - 16)];
    }
    if (k < 96) {
      int j = k - 48;
      return j < 16 ? e[r * 16 + j] : eh[r * HDIM + (j - 16)];
    }
    if (k < 144) {
      int j = k - 96, b = batch[src[r]];
      return j < 16 ? u[b * 16 + j] : uh[b * HDIM + (j - 16)];
    }
    return 0.f;
  }
};

struct RecNodeB {        // [x_c (48) | agg (32) | u_c[batch] (48)] = 128
  const float *x, *xh, *agg, *u, *uh;
  const int* batch;
  __device__ float operator()(int r, int k) const {
    if (k < 48)  return k < 16 ? x[r * 16 + k] : xh[r * HDIM + (k - 16)];
    if (k < 80)  return agg[r * HDIM + (k - 48)];
    int j = k - 80, b = batch[r];
    return j < 16 ? u[b * 16 + j] : uh[b * HDIM + (j - 16)];
  }
};

struct RecGlobB {        // [e_agg (32) | x_agg (32) | u_c (48) | pad] = 112->128
  const float *eagg, *xagg, *u, *uh;
  __device__ float operator()(int r, int k) const {
    if (k < 32)  return eagg[r * HDIM + k];
    if (k < 64)  return xagg[r * HDIM + (k - 32)];
    if (k < 112) {
      int j = k - 64;
      return j < 16 ? u[r * 16 + j] : uh[r * HDIM + (j - 16)];
    }
    return 0.f;
  }
};

struct AttEdgeB {        // [xh[src]-xh[dst] (32) | eh (32) | uh[b[src]] (32)] = 96
  const float *xh, *eh, *uh;
  const int *src, *dst, *batch;
  __device__ float operator()(int r, int k) const {
    if (k < 32) {
      int s = src[r], d = dst[r];
      return xh[s * HDIM + k] - xh[d * HDIM + k];
    }
    if (k < 64) return eh[r * HDIM + (k - 32)];
    return uh[batch[src[r]] * HDIM + (k - 64)];
  }
};

struct AttNodeB {        // [xh (32) | agg (32) | uh[batch] (32)] = 96
  const float *xh, *agg, *uh;
  const int* batch;
  __device__ float operator()(int r, int k) const {
    if (k < 32) return xh[r * HDIM + k];
    if (k < 64) return agg[r * HDIM + (k - 32)];
    return uh[batch[r] * HDIM + (k - 64)];
  }
};

struct Glob96B {         // [e_agg (32) | x_agg (32) | u*(umul?) (32)] = 96
  const float *eagg, *xagg, *ua, *umul;
  __device__ float operator()(int r, int k) const {
    if (k < 32) return eagg[r * HDIM + k];
    if (k < 64) return xagg[r * HDIM + (k - 32)];
    float v = ua[r * HDIM + (k - 64)];
    if (umul) v *= umul[r * HDIM + (k - 64)];
    return v;
  }
};

struct ConcatB {         // [a (32) | b (32)] = 64  (final MLP)
  const float *a, *b;
  __device__ float operator()(int r, int k) const {
    return k < 32 ? a[r * HDIM + k] : b[r * HDIM + (k - 32)];
  }
};

// ---------------------------------------------------------------------------
// Generic fused 2-layer MLP:  out[M,NOUT] = relu(in[M,K1PAD]*W1 + b1)*W2 + b2
// 128 threads = 4 waves; each wave owns a 16-row tile; hidden width fixed 64.
// ---------------------------------------------------------------------------
template <int K1PAD, int NOUT, class Builder>
__global__ void __launch_bounds__(128)
mlp_kernel(Builder bld,
           const _Float16* __restrict__ w1, const float* __restrict__ b1,
           const _Float16* __restrict__ w2, const float* __restrict__ b2,
           float* __restrict__ out, int M) {
  static_assert(K1PAD % 32 == 0 && NOUT % 16 == 0, "tile shape");
  __shared__ _Float16 s_in[64 * K1PAD];
  __shared__ _Float16 s_hid[64 * 64];

  const int t = threadIdx.x;
  const int rowbase = blockIdx.x * 64;

  __builtin_prefetch(w1, 0, 1);   // global_prefetch_b8 (weights -> L2/L0)
  __builtin_prefetch(w2, 0, 1);

  // Cooperative gather of 64 rows into LDS as f16 (adjacent threads hit
  // adjacent k of the same row -> coalesced within each concat segment).
  for (int i = t; i < 64 * K1PAD; i += 128) {
    int r = i / K1PAD, k = i - r * K1PAD;
    int gr = rowbase + r;
    if (gr >= M) gr = M - 1;                 // clamp tail (stores are masked)
    s_in[i] = (_Float16)bld(gr, k);
  }
  __syncthreads();

  const int wave = t >> 5, lane = t & 31;
  const _Float16* atile = s_in + wave * 16 * K1PAD;

  // ---- layer 1: [16,K1PAD] x [K1PAD,64] ----
  v8f acc[4] = {};
  constexpr int NC1 = K1PAD / 32;
#pragma unroll
  for (int c = 0; c < NC1; ++c) {
    v16h a = load_a_frag(atile, K1PAD, c * 32, lane);
#pragma unroll
    for (int nt = 0; nt < 4; ++nt) {
      v16h b = load_b_frag(w1, c * 4 + nt, lane);
      acc[nt] = __builtin_amdgcn_wmma_f32_16x16x32_f16(
          false, a, false, b, (short)0, acc[nt], false, false);
    }
  }

  // bias + relu, transpose D-layout -> row-major f16 via LDS
  _Float16* htile = s_hid + wave * 16 * 64;
  const int nlo = lane & 15;
  const int mh  = (lane >> 4) << 3;          // 0 or 8
#pragma unroll
  for (int nt = 0; nt < 4; ++nt) {
    int n = nt * 16 + nlo;
    float bias = b1[n];
#pragma unroll
    for (int r8 = 0; r8 < 8; ++r8) {
      float v = acc[nt][r8] + bias;
      v = v > 0.f ? v : 0.f;
      htile[(mh + r8) * 64 + n] = (_Float16)v;  // ds_store_b16
    }
  }
  __syncthreads();

  // ---- layer 2: [16,64] x [64,NOUT] ----
  constexpr int NT2 = NOUT / 16;
  v8f acc2[NT2] = {};
#pragma unroll
  for (int c = 0; c < 2; ++c) {
    v16h a = load_a_frag(htile, 64, c * 32, lane);
#pragma unroll
    for (int nt = 0; nt < NT2; ++nt) {
      v16h b = load_b_frag(w2, c * NT2 + nt, lane);
      acc2[nt] = __builtin_amdgcn_wmma_f32_16x16x32_f16(
          false, a, false, b, (short)0, acc2[nt], false, false);
    }
  }

#pragma unroll
  for (int nt = 0; nt < NT2; ++nt) {
    int n = nt * 16 + nlo;
    float bias = b2[n];
#pragma unroll
    for (int r8 = 0; r8 < 8; ++r8) {
      int m = rowbase + wave * 16 + mh + r8;
      if (m < M) out[(size_t)m * NOUT + n] = acc2[nt][r8] + bias;
    }
  }
}

// ---------------------------------------------------------------------------
// Weight pre-swizzle: fp32 [K,N] -> f16 B-fragments (one frag = 512 halfs,
// lane-contiguous: frag f = (kchunk*NT + ntile); lane holds K = c*32 +
// (lane>=16?8:0) + (q&7) + (q>=8?16:0)  (q=0..15), N = ntile*16 + (lane&15)).
// ---------------------------------------------------------------------------
__global__ void prep_weights(const float* __restrict__ W, int K, int Kpad,
                             int N, _Float16* __restrict__ out) {
  int i = blockIdx.x * blockDim.x + threadIdx.x;
  int NT = N >> 4;
  int total = (Kpad >> 5) * NT * 512;
  if (i >= total) return;
  int f = i >> 9, e = i & 511;
  int lane = e >> 4, q = e & 15;
  int c = f / NT, nt = f - c * NT;
  int k = c * 32 + ((lane >> 4) << 3) + (q & 7) + ((q >> 3) << 4);
  int n = nt * 16 + (lane & 15);
  out[i] = (_Float16)((k < K) ? W[k * N + n] : 0.0f);
}

// ---------------------------------------------------------------------------
// Segment mean pieces (f32 atomics; counts with max(c,1) divide)
// ---------------------------------------------------------------------------
__global__ void fill_zero(float* p, int n) {
  int i = blockIdx.x * blockDim.x + threadIdx.x;
  if (i < n) p[i] = 0.f;
}

__global__ void seg_add_kernel(const float* __restrict__ v,
                               const float* __restrict__ v2,
                               const int* __restrict__ idx,
                               const int* __restrict__ indir,
                               float* acc, float* cnt, int rows) {
  int i = blockIdx.x * blockDim.x + threadIdx.x;
  if (i >= rows * HDIM) return;
  int r = i / HDIM, c = i - r * HDIM;
  int s = idx[r];
  if (indir) s = indir[s];
  float val = v[i];
  if (v2) val *= v2[i];
  atomicAdd(acc + (size_t)s * HDIM + c, val);
  if (c == 0) atomicAdd(cnt + s, 1.0f);
}

__global__ void seg_div_kernel(float* acc, const float* __restrict__ cnt, int n) {
  int i = blockIdx.x * blockDim.x + threadIdx.x;
  if (i >= n * HDIM) return;
  float c = cnt[i / HDIM];
  acc[i] = acc[i] / (c > 1.f ? c : 1.f);
}

// ---------------------------------------------------------------------------
// Host orchestration
// ---------------------------------------------------------------------------
#define MLPL(KP, NOo, mi, bld, outp, M)                                        \
  mlp_kernel<KP, NOo><<<((M) + 63) / 64, 128, 0, stream>>>(                    \
      bld, wf1[mi], B1[mi], wf2[mi], B2[mi], (float*)(outp), (M))

#define ZERO(p, n) fill_zero<<<((n) + 255) / 256, 256, 0, stream>>>((float*)(p), (n))

#define SEGADD(v, v2, idx, ind, acc, cnt, rows)                                \
  seg_add_kernel<<<((rows)*HDIM + 255) / 256, 256, 0, stream>>>(               \
      (const float*)(v), (const float*)(v2), (const int*)(idx),                \
      (const int*)(ind), (float*)(acc), (float*)(cnt), (rows))

#define SEGDIV(acc, cnt, n)                                                    \
  seg_div_kernel<<<((n)*HDIM + 255) / 256, 256, 0, stream>>>(                  \
      (float*)(acc), (const float*)(cnt), (n))

extern "C" void kernel_launch(void* const* d_in, const int* in_sizes, int n_in,
                              void* d_out, int out_size, void* d_ws, size_t ws_size,
                              hipStream_t stream) {
  (void)in_sizes; (void)n_in; (void)out_size; (void)ws_size;

  // params pytree flattened with dict keys sorted (jax convention), each MLP
  // contributing (W1, b1, W2, b2):
  enum { AGG = 0, ATT_E, ATT_U, ATT_X, ENC_E, ENC_U, ENC_X, FINAL, REC_E, REC_U, REC_X };
  static const int K1r[11] = {96, 96, 96, 96, 16, 16, 16, 64, 144, 112, 128};
  static const int K1p[11] = {96, 96, 96, 96, 32, 32, 32, 64, 160, 128, 128};
  static const int NOo[11] = {32, 32, 32, 32, 32, 32, 32, 64, 32, 32, 32};

  const float *W1[11], *B1[11], *W2[11], *B2[11];
  for (int i = 0; i < 11; ++i) {
    W1[i] = (const float*)d_in[i * 4 + 0];
    B1[i] = (const float*)d_in[i * 4 + 1];
    W2[i] = (const float*)d_in[i * 4 + 2];
    B2[i] = (const float*)d_in[i * 4 + 3];
  }

  // workspace carve-out
  size_t off = 0;
  char* base = (char*)d_ws;
  auto alloc = [&](size_t bytes) -> void* {
    off = (off + 255) & ~(size_t)255;
    void* p = base + off;
    off += bytes;
    return p;
  };

  // pre-swizzled f16 weights (re-done every call: deterministic, tiny)
  _Float16 *wf1[11], *wf2[11];
  for (int i = 0; i < 11; ++i) {
    int n1 = (K1p[i] / 32) * 4 * 512;
    int n2 = 2 * (NOo[i] / 16) * 512;
    wf1[i] = (_Float16*)alloc((size_t)n1 * 2);
    wf2[i] = (_Float16*)alloc((size_t)n2 * 2);
    prep_weights<<<(n1 + 255) / 256, 256, 0, stream>>>(W1[i], K1r[i], K1p[i], 64, wf1[i]);
    prep_weights<<<(n2 + 255) / 256, 256, 0, stream>>>(W2[i], 64, 64, NOo[i], wf2[i]);
  }

  // graph-state buffers (shared between the two graphs, processed serially)
  float* xh0 = (float*)alloc((size_t)NNODES * HDIM * 4);
  float* xh1 = (float*)alloc((size_t)NNODES * HDIM * 4);
  float* eh0 = (float*)alloc((size_t)NEDGES * HDIM * 4);
  float* eh1 = (float*)alloc((size_t)NEDGES * HDIM * 4);
  float* uh0 = (float*)alloc((size_t)NBATCH * HDIM * 4);
  float* uh1 = (float*)alloc((size_t)NBATCH * HDIM * 4);
  float* ea  = (float*)alloc((size_t)NEDGES * HDIM * 4);
  float* xa  = (float*)alloc((size_t)NNODES * HDIM * 4);
  float* ua  = (float*)alloc((size_t)NBATCH * HDIM * 4);
  float* aggn  = (float*)alloc((size_t)NNODES * HDIM * 4);
  float* cntn  = (float*)alloc((size_t)NNODES * 4);
  float* eaggb = (float*)alloc((size_t)NBATCH * HDIM * 4);
  float* xaggb = (float*)alloc((size_t)NBATCH * HDIM * 4);
  float* cntb  = (float*)alloc((size_t)NBATCH * 4);
  float* cntb2 = (float*)alloc((size_t)NBATCH * 4);
  float* outs  = (float*)alloc((size_t)2 * 2 * NBATCH * HDIM * 4);  // [graph][round][B,32]

  for (int g = 0; g < 2; ++g) {
    const float* XF  = (const float*)d_in[g ? 47 : 44];
    const float* EF  = (const float*)d_in[g ? 48 : 45];
    const float* UF  = (const float*)d_in[g ? 49 : 46];
    const int*   EI  = (const int*)d_in[g ? 52 : 50];
    const int*   BAT = (const int*)d_in[g ? 53 : 51];
    const int* SRC = EI;
    const int* DST = EI + NEDGES;

    float *xhc = xh0, *xhn = xh1;
    float *ehc = eh0, *ehn = eh1;
    float *uhc = uh0, *uhn = uh1;

    // encoders
    MLPL(32, 32, ENC_X, (EncB{XF}), xhc, NNODES);
    MLPL(32, 32, ENC_E, (EncB{EF}), ehc, NEDGES);
    MLPL(32, 32, ENC_U, (EncB{UF}), uhc, NBATCH);

    for (int r = 0; r < 2; ++r) {
      // ---- recurrent MetaLayer ----
      MLPL(160, 32, REC_E, (RecEdgeB{XF, xhc, EF, ehc, UF, uhc, SRC, DST, BAT}),
           ehn, NEDGES);

      ZERO(aggn, NNODES * HDIM); ZERO(cntn, NNODES);
      SEGADD(ehn, nullptr, DST, nullptr, aggn, cntn, NEDGES);
      SEGDIV(aggn, cntn, NNODES);

      MLPL(128, 32, REC_X, (RecNodeB{XF, xhc, aggn, UF, uhc, BAT}), xhn, NNODES);

      ZERO(eaggb, NBATCH * HDIM); ZERO(cntb, NBATCH);
      SEGADD(ehn, nullptr, SRC, BAT, eaggb, cntb, NEDGES);
      SEGDIV(eaggb, cntb, NBATCH);
      ZERO(xaggb, NBATCH * HDIM); ZERO(cntb2, NBATCH);
      SEGADD(xhn, nullptr, BAT, nullptr, xaggb, cntb2, NNODES);
      SEGDIV(xaggb, cntb2, NBATCH);

      MLPL(128, 32, REC_U, (RecGlobB{eaggb, xaggb, UF, uhc}), uhn, NBATCH);

      { float* t;
        t = xhc; xhc = xhn; xhn = t;
        t = ehc; ehc = ehn; ehn = t;
        t = uhc; uhc = uhn; uhn = t; }

      // ---- attention MetaLayer ----
      MLPL(96, 32, ATT_E, (AttEdgeB{xhc, ehc, uhc, SRC, DST, BAT}), ea, NEDGES);

      ZERO(aggn, NNODES * HDIM); ZERO(cntn, NNODES);
      SEGADD(ea, nullptr, DST, nullptr, aggn, cntn, NEDGES);
      SEGDIV(aggn, cntn, NNODES);

      MLPL(96, 32, ATT_X, (AttNodeB{xhc, aggn, uhc, BAT}), xa, NNODES);

      ZERO(eaggb, NBATCH * HDIM); ZERO(cntb, NBATCH);
      SEGADD(ea, nullptr, SRC, BAT, eaggb, cntb, NEDGES);
      SEGDIV(eaggb, cntb, NBATCH);
      ZERO(xaggb, NBATCH * HDIM); ZERO(cntb2, NBATCH);
      SEGADD(xhc, nullptr, BAT, nullptr, xaggb, cntb2, NNODES);
      SEGDIV(xaggb, cntb2, NBATCH);

      MLPL(96, 32, ATT_U, (Glob96B{eaggb, xaggb, uhc, nullptr}), ua, NBATCH);

      // ---- attention-weighted aggregation -> per-round graph embedding ----
      ZERO(eaggb, NBATCH * HDIM); ZERO(cntb, NBATCH);
      SEGADD(ehc, ea, SRC, BAT, eaggb, cntb, NEDGES);    // mean(e_h * e_a)
      SEGDIV(eaggb, cntb, NBATCH);
      ZERO(xaggb, NBATCH * HDIM); ZERO(cntb2, NBATCH);
      SEGADD(xhc, xa, BAT, nullptr, xaggb, cntb2, NNODES);  // mean(x_h * x_a)
      SEGDIV(xaggb, cntb2, NBATCH);

      MLPL(96, 32, AGG, (Glob96B{eaggb, xaggb, uhc, ua}),
           outs + ((size_t)g * 2 + r) * NBATCH * HDIM, NBATCH);
    }
  }

  // final MLP: concat(graph1 emb, graph2 emb) -> [2, B, 64]
  for (int r = 0; r < 2; ++r) {
    MLPL(64, 64, FINAL,
         (ConcatB{outs + (size_t)r * NBATCH * HDIM,
                  outs + (size_t)(2 + r) * NBATCH * HDIM}),
         (float*)d_out + (size_t)r * NBATCH * 64, NBATCH);
  }
}
